// ESA_66958540144912
// MI455X (gfx1250) — matrix operand order repaired and checked by reading, not compiled, not run on hardware
//
#include <hip/hip_runtime.h>

// Problem constants (from the reference)
#define Bn   64
#define Mn   256
#define Dn   256
#define Hn   8
#define DHn  32
#define En   (Bn * Mn)          // 16384 edges total

typedef __attribute__((ext_vector_type(16))) _Float16 v16h;
typedef __attribute__((ext_vector_type(8)))  _Float16 h8;
typedef __attribute__((ext_vector_type(8)))  float    v8f;
typedef __attribute__((ext_vector_type(2)))  _Float16 h2;

__device__ __forceinline__ v8f wmma_f16(v16h a, v16h b, v8f c) {
  // D = A(16x32 f16) * B(32x16 f16) + C(16x16 f32)
  return __builtin_amdgcn_wmma_f32_16x16x32_f16(false, a, false, b,
                                                (short)0, c, false, false);
}

__device__ __forceinline__ v16h cat16(h8 lo, h8 hi) {
  return __builtin_shufflevector(lo, hi, 0, 1, 2, 3, 4, 5, 6, 7,
                                 8, 9, 10, 11, 12, 13, 14, 15);
}

// A-fragment (16x32 f16) from a row-major source. Per ISA layout:
// VGPR0-3 hold K = 8*half + 0..7, VGPR4-7 hold K = 16 + 8*half + 0..7
// -> two contiguous 16B loads per lane.
__device__ __forceinline__ v16h frag_a_rowmajor(const _Float16* p, int lhalf) {
  h8 lo = *(const h8*)(p + lhalf * 8);
  h8 hi = *(const h8*)(p + 16 + lhalf * 8);
  return cat16(lo, hi);
}

// B-fragment (32x16 f16) from a column-major source (per lane, the K run of
// its column is contiguous): VGPR0-3 hold K = 8*half + 0..7, VGPR4-7 hold
// K = 16 + 8*half + 0..7 -> two 16B loads per lane.
__device__ __forceinline__ v16h frag_b_colmajor(const _Float16* p, int lhalf) {
  h8 lo = *(const h8*)(p + lhalf * 8);
  h8 hi = *(const h8*)(p + 16 + lhalf * 8);
  return cat16(lo, hi);
}

// ---------------------------------------------------------------------------
// 1a) f32 -> f16 conversion (row-major preserved), for X
// ---------------------------------------------------------------------------
__global__ void cvt_f32_f16(const float* __restrict__ src,
                            _Float16* __restrict__ dst, int n) {
  int i = blockIdx.x * blockDim.x + threadIdx.x;
  if (i < n) dst[i] = (_Float16)src[i];
}

// ---------------------------------------------------------------------------
// 1b) f32 -> f16 with transpose, for the 256x256 weights:
//     dst[c*256 + r] = src[r*256 + c]  (column-major f16 weights)
// ---------------------------------------------------------------------------
__global__ void cvt_w_transpose(const float* __restrict__ src,
                                _Float16* __restrict__ dst) {
  const int r = blockIdx.x;      // 0..255
  const int c = threadIdx.x;     // 0..255
  dst[c * Dn + r] = (_Float16)src[r * Dn + c];
}

// ---------------------------------------------------------------------------
// 2) Edge-adjacency bitmask: one block per graph (256 threads).
// ---------------------------------------------------------------------------
__global__ void build_mask_kernel(const int* __restrict__ ei,  // [2, E]
                                  unsigned* __restrict__ maskbits) {
  __shared__ int ss[Mn];
  __shared__ int tt[Mn];
  const int b   = blockIdx.x;
  const int tid = threadIdx.x;
  ss[tid] = ei[b * Mn + tid];          // src ids of this graph
  tt[tid] = ei[En + b * Mn + tid];     // tgt ids of this graph
  __syncthreads();

  const int si = ss[tid];
  const int ti = tt[tid];
#pragma unroll
  for (int w = 0; w < 8; ++w) {
    unsigned bits = 0u;
#pragma unroll
    for (int jj = 0; jj < 32; ++jj) {
      const int j  = w * 32 + jj;
      const int sj = ss[j];
      const int tj = tt[j];
      bool adj = (si == sj) | (ti == tj) | (si == tj) | (ti == sj);
      if (j == tid) adj = false;       // fill_diagonal_(0)
      bits |= (adj ? 1u : 0u) << jj;
    }
    maskbits[(b * Mn + tid) * 8 + w] = bits;
  }
}

// ---------------------------------------------------------------------------
// 3) WMMA GEMM: C[rows x 256] = A[rows x 256] * W, with W COLUMN-major
//    (Wt[c*256 + k]). Block = 256 threads = 8 waves; block owns a 16-row
//    stripe; each wave computes two 16x16 tiles, 8 x (16x16x32) WMMAs each.
//    Output precision is a compile-time parameter -> branch-free epilogue.
// ---------------------------------------------------------------------------
template <bool STORE_F32>
__global__ void gemm256_kernel(const _Float16* __restrict__ A,
                               const _Float16* __restrict__ Wt,
                               void* __restrict__ outp) {
  __shared__ _Float16 Al[16 * Dn];                 // 8 KB A tile
  const int tid     = threadIdx.x;
  const int rowbase = blockIdx.x * 16;

  // Stage A tile (16 x 256 f16 = 2048 dwords) cooperatively.
  {
    const unsigned* Ag  = (const unsigned*)(A + (size_t)rowbase * Dn);
    unsigned*       AlU = (unsigned*)Al;
#pragma unroll
    for (int i = 0; i < 8; ++i) AlU[tid + i * 256] = Ag[tid + i * 256];
  }
  __syncthreads();

  const int wave  = tid >> 5;
  const int lane  = tid & 31;
  const int lrow  = lane & 15;   // row within A-frag / col within B-frag
  const int lhalf = lane >> 4;   // lane half selects K sub-range

#pragma unroll
  for (int ct = 0; ct < 2; ++ct) {
    const int col0 = (wave * 2 + ct) * 16;
    v8f acc = {};
#pragma unroll
    for (int kc = 0; kc < 8; ++kc) {
      const int k0 = kc * 32;
      v16h a = frag_a_rowmajor(&Al[lrow * Dn + k0], lhalf);
      v16h b = frag_b_colmajor(&Wt[(size_t)(col0 + lrow) * Dn + k0], lhalf);
      acc = wmma_f16(a, b, acc);
    }
    // C layout: VGPR r -> row (r + 8*half), col = lane%16
    const int colo = col0 + lrow;
#pragma unroll
    for (int r = 0; r < 8; ++r) {
      const int row = rowbase + r + (lhalf << 3);
      if constexpr (STORE_F32)
        ((float*)outp)[(size_t)row * Dn + colo] = acc[r];
      else
        ((_Float16*)outp)[(size_t)row * Dn + colo] = (_Float16)acc[r];
    }
  }
}

// ---------------------------------------------------------------------------
// 4) Masked attention per (graph b, head h). Block = 128 threads = 4 waves.
//    Low-register two-pass softmax: masked scores are spilled tile-by-tile to
//    a wave-private LDS buffer as f16, re-read for exp+sum, and the 1/sum
//    normalization is folded into the output epilogue of the attn@V WMMAs.
//    Score-tile loops are deliberately NOT unrolled (caps register pressure);
//    launch_bounds(128,1) lets the backend use a large VGPR budget.
// ---------------------------------------------------------------------------
__global__ void __launch_bounds__(128, 1)
attn_kernel(const _Float16* __restrict__ Qh,
            const _Float16* __restrict__ Kh,
            const _Float16* __restrict__ Vh,
            const unsigned* __restrict__ maskbits,
            _Float16* __restrict__ Oh) {
  __shared__ unsigned maskL[Mn * 8];          //  8 KB
  __shared__ _Float16 Vt[DHn * Mn];           // 16 KB  (Vt[d][e], d=0..31)
  __shared__ _Float16 probs[4][16 * Mn];      // 32 KB (per-wave 16x256)

  const int b     = blockIdx.x >> 3;
  const int h     = blockIdx.x & 7;
  const int tid   = threadIdx.x;
  const int wave  = tid >> 5;
  const int lane  = tid & 31;
  const int lrow  = lane & 15;
  const int lhalf = lane >> 4;

  // Stage adjacency bits (2048 dwords) and V[b,:,h,:]^T (32 x 256 f16).
  {
    const unsigned* mg = maskbits + (size_t)b * Mn * 8;
#pragma unroll
    for (int i = 0; i < 16; ++i) maskL[tid + i * 128] = mg[tid + i * 128];
    for (int e = tid; e < Mn; e += 128) {
      const h2* src = (const h2*)(Vh + ((size_t)(b * Mn + e)) * Dn + h * DHn);
#pragma unroll
      for (int j = 0; j < 16; ++j) {
        h2 p = src[j];
        Vt[(2 * j) * Mn + e]     = p[0];
        Vt[(2 * j + 1) * Mn + e] = p[1];
      }
    }
  }
  __syncthreads();

  const float scale  = 0.17677669529663687f;  // 1/sqrt(32)
  const float NEGF16 = -60000.0f;             // finite in f16; exp -> 0
  const _Float16* Kbase = Kh + ((size_t)(b * Mn)) * Dn + h * DHn;
  _Float16* pw = &probs[wave][0];             // wave-private 16x256 buffer

  // 16 row-tiles of 16 query rows; 4 waves -> 4 tiles each.
  for (int rt = wave; rt < 16; rt += 4) {
    const int qbase = rt * 16;

    // Q A-fragment: 16 rows x 32 K (single chunk, dh = 32); two 16B loads.
    v16h aq = frag_a_rowmajor(
        Qh + ((size_t)(b * Mn + qbase + lrow)) * Dn + h * DHn, lhalf);

    // Pass A: score tiles -> scale+mask -> running row max -> spill f16 to LDS
    float mx[8];
#pragma unroll
    for (int r = 0; r < 8; ++r) mx[r] = -1e30f;
#pragma unroll 1
    for (int t = 0; t < 16; ++t) {
      // B = K^T: lane's column is edge (t*16+lrow); its K-run is the
      // contiguous d-range of Kh's row -> column-major pattern.
      v16h bk = frag_b_colmajor(Kbase + (size_t)(t * 16 + lrow) * Dn, lhalf);
      v8f c = {};
      c = wmma_f16(aq, bk, c);
      const int bit = ((t & 1) << 4) | lrow;   // column bit within mask word
#pragma unroll
      for (int r = 0; r < 8; ++r) {
        const int row = r + (lhalf << 3);      // local row in tile (C layout)
        const unsigned wbits = maskL[(qbase + row) * 8 + (t >> 1)];
        const float val = ((wbits >> bit) & 1u) ? c[r] * scale : NEGF16;
        mx[r] = fmaxf(mx[r], val);
        pw[row * Mn + t * 16 + lrow] = (_Float16)val;
      }
    }
    // Row max across the 16 lanes that share a row.
#pragma unroll
    for (int r = 0; r < 8; ++r)
      for (int o = 1; o < 16; o <<= 1)
        mx[r] = fmaxf(mx[r], __shfl_xor(mx[r], o, 16));

    // Pass B: exp + row sum; store UNNORMALIZED exp back to LDS.
    float sum[8];
#pragma unroll
    for (int r = 0; r < 8; ++r) {
      const int row = r + (lhalf << 3);
      float acc = 0.0f;
#pragma unroll 1
      for (int t = 0; t < 16; ++t) {
        const float sv = (float)pw[row * Mn + t * 16 + lrow];
        const float e  = __expf(sv - mx[r]);
        acc += e;
        pw[row * Mn + t * 16 + lrow] = (_Float16)e;
      }
      for (int o = 1; o < 16; o <<= 1) acc += __shfl_xor(acc, o, 16);
      sum[r] = acc;
    }

    // attn @ V with unnormalized probs: 16 x 32 output, 8 K-chunks of 32.
    v8f o0 = {}, o1 = {};
#pragma unroll
    for (int ec = 0; ec < 8; ++ec) {
      v16h ap = frag_a_rowmajor(&pw[lrow * Mn + ec * 32], lhalf);
#pragma unroll
      for (int ct = 0; ct < 2; ++ct) {
        // B = V: lane's column is feature dd; its K(e)-run is contiguous in Vt
        v16h bv = frag_b_colmajor(&Vt[(ct * 16 + lrow) * Mn + ec * 32], lhalf);
        if (ct == 0) o0 = wmma_f16(ap, bv, o0);
        else         o1 = wmma_f16(ap, bv, o1);
      }
    }

    // Normalize by 1/rowsum in the epilogue and store O[b, row, h, :] as f16.
#pragma unroll
    for (int r = 0; r < 8; ++r) {
      const float inv = 1.0f / sum[r];
      const int row = qbase + r + (lhalf << 3);
      _Float16* op = Oh + ((size_t)(b * Mn + row)) * Dn + h * DHn;
      op[lrow]      = (_Float16)(o0[r] * inv);
      op[16 + lrow] = (_Float16)(o1[r] * inv);
    }
  }
}

// ---------------------------------------------------------------------------
// Launcher
// ---------------------------------------------------------------------------
extern "C" void kernel_launch(void* const* d_in, const int* in_sizes, int n_in,
                              void* d_out, int out_size, void* d_ws,
                              size_t ws_size, hipStream_t stream) {
  const float* X  = (const float*)d_in[0];
  const float* Wq = (const float*)d_in[1];
  const float* Wk = (const float*)d_in[2];
  const float* Wv = (const float*)d_in[3];
  const float* Wo = (const float*)d_in[4];
  const int*   ei = (const int*)d_in[5];

  // Workspace layout (all offsets 256B-aligned by construction)
  char* ws = (char*)d_ws;
  const size_t XH  = (size_t)En * Dn * sizeof(_Float16);   // 8 MB
  const size_t WHS = (size_t)Dn * Dn * sizeof(_Float16);   // 128 KB each
  _Float16* Xh = (_Float16*)ws;
  _Float16* Wh = (_Float16*)(ws + XH);                     // Wq|Wk|Wv|Wo (col-major)
  _Float16* Qh = (_Float16*)(ws + XH + 4 * WHS);
  _Float16* Kh = (_Float16*)(ws + XH + 4 * WHS + 1 * XH);
  _Float16* Vh = (_Float16*)(ws + XH + 4 * WHS + 2 * XH);
  _Float16* Oh = (_Float16*)(ws + XH + 4 * WHS + 3 * XH);
  unsigned* maskbits = (unsigned*)(ws + XH + 4 * WHS + 4 * XH);

  const int nW = Dn * Dn;
  cvt_f32_f16<<<(En * Dn + 255) / 256, 256, 0, stream>>>(X, Xh, En * Dn);
  cvt_w_transpose<<<Dn, Dn, 0, stream>>>(Wq, Wh + 0 * nW);
  cvt_w_transpose<<<Dn, Dn, 0, stream>>>(Wk, Wh + 1 * nW);
  cvt_w_transpose<<<Dn, Dn, 0, stream>>>(Wv, Wh + 2 * nW);
  cvt_w_transpose<<<Dn, Dn, 0, stream>>>(Wo, Wh + 3 * nW);

  build_mask_kernel<<<Bn, Mn, 0, stream>>>(ei, maskbits);

  // QKV projections (f16 out, f32 accumulate inside WMMA)
  gemm256_kernel<false><<<En / 16, 256, 0, stream>>>(Xh, Wh + 0 * nW, Qh);
  gemm256_kernel<false><<<En / 16, 256, 0, stream>>>(Xh, Wh + 1 * nW, Kh);
  gemm256_kernel<false><<<En / 16, 256, 0, stream>>>(Xh, Wh + 2 * nW, Vh);

  // Masked attention per (graph, head)
  attn_kernel<<<Bn * Hn, 128, 0, stream>>>(Qh, Kh, Vh, maskbits, Oh);

  // Output projection straight to f32 result
  gemm256_kernel<true><<<En / 16, 256, 0, stream>>>(Oh, Wh + 3 * nW, d_out);
}
